// DualConsistencyRegularization_28046136443239
// MI455X (gfx1250) — matrix operand order repaired and checked by reading, not compiled, not run on hardware
//
#include <hip/hip_runtime.h>

#define BB 64
#define SS 915
#define DD 256
#define KK 30
#define HH 128
#define LL 30
#define ROWS (3 * BB * KK) /* 5760 */
#define EPSLN 1e-5f

typedef _Float16 v16h __attribute__((ext_vector_type(16)));
typedef float v8f __attribute__((ext_vector_type(8)));

// ---------------- block-wide sum (wave32 shuffle + LDS across waves) ----------------
__device__ __forceinline__ float block_sum(float v, float* sm) {
#pragma unroll
  for (int off = 16; off > 0; off >>= 1) v += __shfl_down(v, off, 32);
  int lane = threadIdx.x & 31;
  int wid = threadIdx.x >> 5;
  if (lane == 0) sm[wid] = v;
  __syncthreads();
  int nw = blockDim.x >> 5;
  float r = (threadIdx.x < (unsigned)nw) ? sm[threadIdx.x] : 0.0f;
  if (wid == 0) {
#pragma unroll
    for (int off = 16; off > 0; off >>= 1) r += __shfl_down(r, off, 32);
    if (lane == 0) sm[0] = r;
  }
  __syncthreads();
  float out = sm[0];
  __syncthreads();
  return out;
}

// ---------------- wave32 all-lanes sum via xor butterfly ----------------
__device__ __forceinline__ float wave_sum(float v) {
#pragma unroll
  for (int off = 16; off > 0; off >>= 1) v += __shfl_xor(v, off, 32);
  return v;
}

// ---------------- pack f32 weight [K,N] into WMMA B-fragment order (f16) ----------------
// dst layout: ((kc * (N/16) + nt) * 32 + lane) * 16 + r, lane = half*16 + (n&15),
// r = local K row within half (0..15). Each lane then reads 16 contiguous f16 (2x b128).
__global__ void pack_b(const float* __restrict__ W, _Float16* __restrict__ Bp, int Kdim,
                       int Ndim) {
  int i = blockIdx.x * blockDim.x + threadIdx.x;
  if (i >= Kdim * Ndim) return;
  int krow = i / Ndim, n = i % Ndim;
  int kc = krow >> 5;
  int h = (krow >> 4) & 1;
  int r = krow & 15;
  int nt = n >> 4, nl = n & 15;
  int lane = h * 16 + nl;
  size_t dst = (((size_t)kc * (Ndim >> 4) + nt) * 32 + lane) * 16 + r;
  Bp[dst] = (_Float16)W[i];
}

// ---------------- fused token-LN + segment mean pooling ----------------
// one block per (stream, b, k); 8 waves; each wave LNs whole tokens (lane = 8 dims, b128 loads),
// reductions are wave-internal shuffles (no barriers in the token loop).
__global__ void ln_segpool(const float* __restrict__ x0, const float* __restrict__ x1,
                           const float* __restrict__ x2, const float* __restrict__ w0,
                           const float* __restrict__ b0, const float* __restrict__ w1,
                           const float* __restrict__ b1, const float* __restrict__ w2,
                           const float* __restrict__ b2, _Float16* __restrict__ seg) {
  __shared__ float accs[8 * DD];
  int blk = blockIdx.x; // == (st*BB + b)*KK + k
  int st = blk / (BB * KK);
  int rem = blk % (BB * KK);
  int b = rem / KK;
  int k = rem % KK;
  const float* x = (st == 0) ? x0 : (st == 1 ? x1 : x2);
  const float* gw = (st == 0) ? w0 : (st == 1 ? w1 : w2);
  const float* gb = (st == 0) ? b0 : (st == 1 ? b1 : b2);
  int start = k * LL;
  int len = (k == KK - 1) ? (SS - start) : LL;
  int lane = threadIdx.x & 31;
  int wid = threadIdx.x >> 5;
  int d0 = lane * 8;

  float g[8], be[8], acc[8];
  {
    const float4* g4 = (const float4*)(gw + d0);
    const float4* b4 = (const float4*)(gb + d0);
    float4 ga = g4[0], gbv = g4[1], ba = b4[0], bbv = b4[1];
    g[0] = ga.x; g[1] = ga.y; g[2] = ga.z; g[3] = ga.w;
    g[4] = gbv.x; g[5] = gbv.y; g[6] = gbv.z; g[7] = gbv.w;
    be[0] = ba.x; be[1] = ba.y; be[2] = ba.z; be[3] = ba.w;
    be[4] = bbv.x; be[5] = bbv.y; be[6] = bbv.z; be[7] = bbv.w;
  }
#pragma unroll
  for (int i = 0; i < 8; ++i) acc[i] = 0.0f;

  for (int t = wid; t < len; t += 8) {
    const float4* row = (const float4*)(x + ((size_t)b * SS + start + t) * DD + d0);
    float4 v0 = row[0], v1 = row[1];
    float v[8] = {v0.x, v0.y, v0.z, v0.w, v1.x, v1.y, v1.z, v1.w};
    float s = v[0] + v[1] + v[2] + v[3] + v[4] + v[5] + v[6] + v[7];
    float mu = wave_sum(s) * (1.0f / DD);
    float sq = 0.0f;
#pragma unroll
    for (int i = 0; i < 8; ++i) {
      float df = v[i] - mu;
      v[i] = df;
      sq += df * df;
    }
    float rstd = rsqrtf(wave_sum(sq) * (1.0f / DD) + EPSLN);
#pragma unroll
    for (int i = 0; i < 8; ++i) acc[i] += v[i] * rstd * g[i] + be[i];
  }
#pragma unroll
  for (int i = 0; i < 8; ++i) accs[wid * DD + d0 + i] = acc[i];
  __syncthreads();
  int d = threadIdx.x; // 0..255
  float tot = 0.0f;
#pragma unroll
  for (int w = 0; w < 8; ++w) tot += accs[w * DD + d];
  seg[(size_t)blk * DD + d] = (_Float16)(tot / (float)len);
}

// ---------------- WMMA GEMM: C[M,N] = A[M,KDIM] * Bp + bias ----------------
// A row-major f16; Bp pre-packed fragment order. Block = 128 threads (4 waves),
// each wave -> one 16x16 tile. grid = (M/16, N/64). Fully unrolled k loop.
template <int KDIM>
__global__ void gemm_wmma(const _Float16* __restrict__ A, const _Float16* __restrict__ Bp,
                          const float* __restrict__ bias, float* __restrict__ C, int Ndim) {
  int wid = threadIdx.x >> 5;
  int lane = threadIdx.x & 31;
  int m0 = blockIdx.x << 4;
  int ntile = (blockIdx.y << 2) + wid;
  int n0 = ntile << 4;
  int ml = lane & 15;
  int h = lane >> 4;
  int ntiles = Ndim >> 4;
  v8f c = {0.f, 0.f, 0.f, 0.f, 0.f, 0.f, 0.f, 0.f};
  const uint4* arow = (const uint4*)(A + (size_t)(m0 + ml) * KDIM);
#pragma unroll
  for (int kc = 0; kc < KDIM / 32; ++kc) {
    union { v16h v; uint4 q[2]; } au, bu;
    // A fragment: words [8h..8h+7] of this 32-element chunk (k = 16h + s)
    au.q[0] = arow[kc * 4 + 2 * h];
    au.q[1] = arow[kc * 4 + 2 * h + 1];
    // B fragment: 16 contiguous f16 per lane from packed buffer
    const uint4* bp = (const uint4*)(Bp + (((size_t)kc * ntiles + ntile) * 32 + lane) * 16);
    bu.q[0] = bp[0];
    bu.q[1] = bp[1];
    c = __builtin_amdgcn_wmma_f32_16x16x32_f16(false, au.v, false, bu.v, (short)0, c, false,
                                               false);
  }
  float bv = bias ? bias[n0 + ml] : 0.0f;
#pragma unroll
  for (int r = 0; r < 8; ++r) {
    int mm = m0 + r + 8 * h; // C layout: VGPR r -> M = r + 8*half, N = lane&15
    C[(size_t)mm * Ndim + n0 + ml] = c[r] + bv;
  }
}

// ---------------- row LayerNorm (+optional ReLU), f32 in, f32/f16 out ----------------
__global__ void rowln(const float* __restrict__ in, float* __restrict__ of32,
                      _Float16* __restrict__ of16, const float* __restrict__ g,
                      const float* __restrict__ be, int N, int relu) {
  __shared__ float sm[32];
  size_t row = blockIdx.x;
  int t = threadIdx.x;
  float v = in[row * N + t];
  float mu = block_sum(v, sm) / (float)N;
  float df = v - mu;
  float var = block_sum(df * df, sm) / (float)N;
  float y = df * rsqrtf(var + EPSLN) * g[t] + be[t];
  if (relu) y = fmaxf(y, 0.0f);
  if (of32) of32[row * N + t] = y;
  if (of16) of16[row * N + t] = (_Float16)y;
}

// ---------------- per-(stream,b) GCN: sim -> softmax(time_att) -> A_norm @ (enc@W) ----------------
__global__ void gcn_kernel(const float* __restrict__ enc, const float* __restrict__ encg,
                           float* __restrict__ upd) {
  __shared__ float sE[KK * HH];
  __shared__ float sG[KK * HH];
  __shared__ float sS[KK * 32];
  __shared__ float sDeg[KK];
  int base = blockIdx.x * KK; // row offset into [ROWS, HH]
  int tid = threadIdx.x;
  for (int i = tid; i < KK * HH; i += blockDim.x) {
    sE[i] = enc[(size_t)base * HH + i];
    sG[i] = encg[(size_t)base * HH + i];
  }
  __syncthreads();
  for (int e = tid; e < KK * KK; e += blockDim.x) {
    int k = e / KK, j = e % KK;
    float s = 0.0f;
#pragma unroll 4
    for (int hh = 0; hh < HH; ++hh) s += sE[k * HH + hh] * sE[j * HH + hh];
    sS[k * 32 + j] = s * 0.08838834764831845f; // 1/sqrt(128)
  }
  __syncthreads();
  if (tid < KK) {
    int k = tid;
    float tmp[KK];
    float mx = -1e30f;
    for (int j = 0; j < KK; ++j) {
      float att = __expf(-0.1f * fabsf((float)(k - j)));
      float v = sS[k * 32 + j] * att;
      tmp[j] = v;
      mx = fmaxf(mx, v);
    }
    float sum = 0.0f;
    for (int j = 0; j < KK; ++j) {
      float e2 = __expf(tmp[j] - mx);
      tmp[j] = e2;
      sum += e2;
    }
    float inv = 1.0f / sum;
    float rs = 0.0f;
    for (int j = 0; j < KK; ++j) {
      float a = tmp[j] * inv + ((j == k) ? 1.0f : 0.0f);
      sS[k * 32 + j] = a;
      rs += a;
    }
    sDeg[k] = rsqrtf(rs);
  }
  __syncthreads();
  for (int e = tid; e < KK * HH; e += blockDim.x) {
    int k = e >> 7, hh = e & (HH - 1);
    float dk = sDeg[k];
    float acc = 0.0f;
#pragma unroll
    for (int j = 0; j < KK; ++j) acc += sS[k * 32 + j] * sDeg[j] * sG[j * HH + hh];
    upd[(size_t)(base + k) * HH + hh] = dk * acc;
  }
}

// ---------------- regularization loss ----------------
__global__ void init_loss(float* loss) {
  if (threadIdx.x == 0 && blockIdx.x == 0) *loss = 0.0f;
}

__global__ void regloss(const float* __restrict__ u, float* __restrict__ loss) {
  __shared__ float sm[32];
  int b = blockIdx.x;
  const float* ud = u + (size_t)b * KK * HH;
  const float* un = u + (size_t)(BB + b) * KK * HH;
  const float* ue = u + (size_t)(2 * BB + b) * KK * HH;
  float self = 0.0f, cross = 0.0f;
  for (int i = threadIdx.x; i < KK * HH; i += blockDim.x) {
    float a = ud[i], c = un[i], e = ue[i];
    self += a * a + c * c + e * e;
    cross += a * c + a * e + c * e;
  }
  self = block_sum(self, sm);
  cross = block_sum(cross, sm);
  if (threadIdx.x == 0) atomicAdd(loss, (self - 0.5f * cross) * (1.0f / BB));
}

// ---------------- scatter: out = 0.9*x + 0.1*proj[seg_id], float4 vectorized ----------------
__global__ void scatter_out(const float4* __restrict__ x0, const float4* __restrict__ x1,
                            const float4* __restrict__ x2, const float4* __restrict__ proj,
                            float4* __restrict__ out) {
  size_t i = (size_t)blockIdx.x * blockDim.x + threadIdx.x;
  size_t total = (size_t)3 * BB * SS * (DD / 4);
  if (i >= total) return;
  int dq = (int)(i & 63); // 64 float4 per row of 256
  size_t r = i >> 6;
  int s = (int)(r % SS);
  r /= SS;
  int b = (int)(r % BB);
  int st = (int)(r / BB);
  const float4* x = (st == 0) ? x0 : (st == 1 ? x1 : x2);
  int seg = s / LL;
  if (seg > KK - 1) seg = KK - 1;
  float4 pv = proj[(((size_t)(st * BB + b)) * KK + seg) * (DD / 4) + dq];
  float4 xv = x[((size_t)b * SS + s) * (DD / 4) + dq];
  float4 o;
  o.x = 0.9f * xv.x + 0.1f * pv.x;
  o.y = 0.9f * xv.y + 0.1f * pv.y;
  o.z = 0.9f * xv.z + 0.1f * pv.z;
  o.w = 0.9f * xv.w + 0.1f * pv.w;
  out[i] = o;
}

// ---------------- launcher ----------------
extern "C" void kernel_launch(void* const* d_in, const int* in_sizes, int n_in, void* d_out,
                              int out_size, void* d_ws, size_t ws_size, hipStream_t stream) {
  (void)in_sizes; (void)n_in; (void)out_size; (void)ws_size;

  const float* x_d = (const float*)d_in[0];
  const float* x_n = (const float*)d_in[1];
  const float* x_e = (const float*)d_in[2];

  char* p = (char*)d_ws;
  auto bump = [&](size_t bytes) -> void* {
    void* r = (void*)p;
    p += (bytes + 255) & ~(size_t)255;
    return r;
  };
  _Float16* segf16 = (_Float16*)bump((size_t)ROWS * DD * 2);
  _Float16* w1p = (_Float16*)bump((size_t)DD * HH * 2);
  _Float16* w2p = (_Float16*)bump((size_t)HH * HH * 2);
  _Float16* gwp = (_Float16*)bump((size_t)HH * HH * 2);
  _Float16* pwp = (_Float16*)bump((size_t)HH * DD * 2);
  float* h1 = (float*)bump((size_t)ROWS * HH * 4);
  _Float16* h1h = (_Float16*)bump((size_t)ROWS * HH * 2);
  float* h2 = (float*)bump((size_t)ROWS * HH * 4);
  float* enc = (float*)bump((size_t)ROWS * HH * 4);
  _Float16* ench = (_Float16*)bump((size_t)ROWS * HH * 2);
  float* encg = (float*)bump((size_t)ROWS * HH * 4);
  float* upd = (float*)bump((size_t)ROWS * HH * 4);
  float* uf = (float*)bump((size_t)ROWS * HH * 4);
  _Float16* uh = (_Float16*)bump((size_t)ROWS * HH * 2);
  float* pr = (float*)bump((size_t)ROWS * DD * 4);
  float* prln = (float*)bump((size_t)ROWS * DD * 4);

  // pack weights into WMMA B-fragment order (tiny, amortized)
  pack_b<<<(DD * HH + 255) / 256, 256, 0, stream>>>((const float*)d_in[9], w1p, DD, HH);
  pack_b<<<(HH * HH + 255) / 256, 256, 0, stream>>>((const float*)d_in[13], w2p, HH, HH);
  pack_b<<<(HH * HH + 255) / 256, 256, 0, stream>>>((const float*)d_in[17], gwp, HH, HH);
  pack_b<<<(HH * DD + 255) / 256, 256, 0, stream>>>((const float*)d_in[20], pwp, HH, DD);

  // fused LN + segment pooling (dominant streaming read of x, b128 loads, shuffle LN)
  ln_segpool<<<3 * BB * KK, DD, 0, stream>>>(
      x_d, x_n, x_e, (const float*)d_in[3], (const float*)d_in[4], (const float*)d_in[5],
      (const float*)d_in[6], (const float*)d_in[7], (const float*)d_in[8], segf16);

  // MLP layer 1: [5760,256] x [256,128]
  gemm_wmma<DD><<<dim3(ROWS / 16, HH / 64), 128, 0, stream>>>(segf16, w1p,
                                                              (const float*)d_in[10], h1, HH);
  rowln<<<ROWS, HH, 0, stream>>>(h1, nullptr, h1h, (const float*)d_in[11],
                                 (const float*)d_in[12], HH, 1);
  // MLP layer 2: [5760,128] x [128,128]
  gemm_wmma<HH><<<dim3(ROWS / 16, HH / 64), 128, 0, stream>>>(h1h, w2p, (const float*)d_in[14],
                                                              h2, HH);
  rowln<<<ROWS, HH, 0, stream>>>(h2, enc, ench, (const float*)d_in[15], (const float*)d_in[16],
                                 HH, 0);
  // enc @ gcn_w: [5760,128] x [128,128]
  gemm_wmma<HH><<<dim3(ROWS / 16, HH / 64), 128, 0, stream>>>(ench, gwp, nullptr, encg, HH);

  // per-(stream,batch) GCN aggregation
  gcn_kernel<<<3 * BB, 256, 0, stream>>>(enc, encg, upd);
  rowln<<<ROWS, HH, 0, stream>>>(upd, uf, uh, (const float*)d_in[18], (const float*)d_in[19],
                                 HH, 1);

  // projection: [5760,128] x [128,256]
  gemm_wmma<HH><<<dim3(ROWS / 16, DD / 64), 128, 0, stream>>>(uh, pwp, (const float*)d_in[21],
                                                              pr, DD);
  rowln<<<ROWS, DD, 0, stream>>>(pr, prln, nullptr, (const float*)d_in[22],
                                 (const float*)d_in[23], DD, 0);

  // reg loss (scalar at end of d_out)
  float* out = (float*)d_out;
  float* loss = out + (size_t)3 * BB * SS * DD;
  init_loss<<<1, 1, 0, stream>>>(loss);
  regloss<<<BB, 256, 0, stream>>>(uf, loss);

  // final scatter (dominant streaming write, b128)
  size_t total4 = (size_t)3 * BB * SS * (DD / 4);
  scatter_out<<<(unsigned)((total4 + 255) / 256), 256, 0, stream>>>(
      (const float4*)x_d, (const float4*)x_n, (const float4*)x_e, (const float4*)prln,
      (float4*)out);
}
